// BeamSearchDecoder_56573309223484
// MI455X (gfx1250) — compile-verified, matrix operand order checked by reading, not run on hardware
//
#include <hip/hip_runtime.h>
#include <hip/hip_bf16.h>
#include <math.h>

// ---------------- problem constants ----------------
#define VSZ    50257
#define HD     256
#define EMBD   256
#define BB     8
#define SS     32
#define BEAMS  4
#define MAXLEN 32
#define RR     32            // B*BEAMS beam rows
#define KK     512           // 2*HD (concat h2, ctx)
#define NT     3142          // ceil(VSZ/16)
#define NPAD   (NT*16)       // 50272
#define NEGF   (-1e9f)

typedef _Float16 f16;
typedef __attribute__((ext_vector_type(16))) _Float16 v16h;
typedef __attribute__((ext_vector_type(8)))  float    v8f;

union VU { v16h v; uint4 q[2]; };

__device__ __forceinline__ float sigf(float x) { return 1.0f / (1.0f + expf(-x)); }

// ---------------- workspace layout (bytes) ----------------
#define OFF_WO16  ((size_t)0)                                  // f16 [NT*16 tiles][512]
#define OFF_LOG   (OFF_WO16 + (size_t)512*NPAD*2)              // f32 [32][NPAD]
#define OFF_ENC   (OFF_LOG  + (size_t)RR*NPAD*4)               // f32 [8][32][256]
#define OFF_H     (OFF_ENC  + (size_t)BB*SS*HD*4)              // f32 [2][32][256]
#define OFF_C     (OFF_H    + (size_t)2*RR*HD*4)
#define OFF_A16   (OFF_C    + (size_t)2*RR*HD*4)               // f16 [32][512]
#define OFF_SEQ0  (OFF_A16  + (size_t)RR*KK*2)                 // i32 [8][4][32]
#define OFF_SEQ1  (OFF_SEQ0 + 4096)
#define OFF_BS0   (OFF_SEQ1 + 4096)                            // f32 [32]
#define OFF_BS1   (OFF_BS0  + 256)
#define OFF_LOGZ  (OFF_BS1  + 256)                             // f32 [32]
#define OFF_TOPI  (OFF_LOGZ + 256)                             // i32 [8][4]
#define OFF_BLK   (OFF_TOPI + 256)                             // i32 [32]

// ============================================================
// 1) Convert Wo (f32 row-major [512][V]) into f16, tiled into the
//    WMMA B-operand lane layout: tile (nt,kt) = 32 lanes x 16 halfs.
//    lane<16 : n=nt*16+lane,    elems 0..7 -> k=k0..k0+7,  8..15 -> k0+16..23
//    lane>=16: n=nt*16+lane-16, elems 0..7 -> k0+8..15,    8..15 -> k0+24..31
// ============================================================
__global__ void k_convert_wo(const float* __restrict__ Wo, f16* __restrict__ W16) {
  int idx = blockIdx.x * 256 + threadIdx.x;
  const int total = NT * 16 * 512;
  if (idx >= total) return;
  int e  = idx & 15;
  int l  = (idx >> 4) & 31;
  int kt = (idx >> 9) & 15;
  int nt = idx >> 13;
  int lo = l & 15;
  int hi = l >> 4;
  int n  = nt * 16 + lo;
  int kk = (e < 8) ? (hi * 8 + e) : (16 + hi * 8 + (e - 8));
  int k  = kt * 32 + kk;
  float v = (n < VSZ) ? Wo[(size_t)k * VSZ + n] : 0.0f;
  W16[(size_t)(nt * 16 + kt) * 512 + l * 16 + e] = (f16)v;
}

// ============================================================
// 2) Encoder: one block per batch row, 256 threads = one H column each.
// ============================================================
__global__ void __launch_bounds__(256) k_encoder(
    const int* __restrict__ src, const float* __restrict__ E,
    const float* __restrict__ Wi1, const float* __restrict__ Wh1, const float* __restrict__ b1,
    const float* __restrict__ Wi2, const float* __restrict__ Wh2, const float* __restrict__ b2,
    float* __restrict__ enc, float* __restrict__ Hb, float* __restrict__ Cb) {
  int b = blockIdx.x;
  int j = threadIdx.x;
  __shared__ float x[HD], h1[HD], h2[HD];
  float c1 = 0.f, c2 = 0.f;
  h1[j] = 0.f; h2[j] = 0.f;
  __syncthreads();
  for (int s = 0; s < SS; ++s) {
    int tok = src[b * SS + s];
    x[j] = E[(size_t)tok * EMBD + j];
    __syncthreads();
    // layer 1
    float z[4];
    for (int g = 0; g < 4; ++g) z[g] = b1[g * HD + j];
    for (int k = 0; k < EMBD; ++k) {
      float xv = x[k];
      for (int g = 0; g < 4; ++g) z[g] += xv * Wi1[(size_t)k * (4*HD) + g * HD + j];
    }
    for (int k = 0; k < HD; ++k) {
      float hv = h1[k];
      for (int g = 0; g < 4; ++g) z[g] += hv * Wh1[(size_t)k * (4*HD) + g * HD + j];
    }
    c1 = sigf(z[1]) * c1 + sigf(z[0]) * tanhf(z[2]);
    float h1v = sigf(z[3]) * tanhf(c1);
    __syncthreads();
    h1[j] = h1v;
    __syncthreads();
    // layer 2
    for (int g = 0; g < 4; ++g) z[g] = b2[g * HD + j];
    for (int k = 0; k < HD; ++k) {
      float xv = h1[k];
      for (int g = 0; g < 4; ++g) z[g] += xv * Wi2[(size_t)k * (4*HD) + g * HD + j];
    }
    for (int k = 0; k < HD; ++k) {
      float hv = h2[k];
      for (int g = 0; g < 4; ++g) z[g] += hv * Wh2[(size_t)k * (4*HD) + g * HD + j];
    }
    c2 = sigf(z[1]) * c2 + sigf(z[0]) * tanhf(z[2]);
    float h2v = sigf(z[3]) * tanhf(c2);
    __syncthreads();
    h2[j] = h2v;
    enc[((size_t)b * SS + s) * HD + j] = h2v;
    __syncthreads();
  }
  // decoder init: tiled replication (row i gets hidden of batch i%8)
  for (int q = 0; q < 4; ++q) {
    int i = b + q * 8;
    Hb[i * HD + j]            = h1[j];
    Hb[RR * HD + i * HD + j]  = h2[j];
    Cb[i * HD + j]            = c1;
    Cb[RR * HD + i * HD + j]  = c2;
  }
}

// ============================================================
// 3) init seq / beam scores
// ============================================================
__global__ void k_init(int* __restrict__ seq0, float* __restrict__ bs0) {
  int tid = blockIdx.x * 256 + threadIdx.x;
  if (tid < RR * MAXLEN) seq0[tid] = ((tid & 31) == 0) ? 1 : 0;
  if (tid < RR) bs0[tid] = ((tid & 3) == 0) ? 0.0f : NEGF;
}

// ============================================================
// 4) decoder step: one block per beam row.  LSTM1, LSTM2, attention,
//    pack A=[h2||ctx] to f16, ngram block token.
// ============================================================
__global__ void __launch_bounds__(256) k_step(
    const int* __restrict__ seq_in, const float* __restrict__ E,
    const float* __restrict__ Wi1, const float* __restrict__ Wh1, const float* __restrict__ b1,
    const float* __restrict__ Wi2, const float* __restrict__ Wh2, const float* __restrict__ b2,
    const float* __restrict__ enc, float* __restrict__ Hb, float* __restrict__ Cb,
    f16* __restrict__ A16, int* __restrict__ blk, int t) {
  int r = blockIdx.x;            // 0..31
  int j = threadIdx.x;           // 0..255
  int b = r >> 2;
  __shared__ float x[HD], h1o[HD], h2o[HD], h1n[HD], h2n[HD], att[SS];
  int tok = seq_in[r * MAXLEN + (t - 1)];
  x[j]   = E[(size_t)tok * EMBD + j];
  h1o[j] = Hb[r * HD + j];
  h2o[j] = Hb[RR * HD + r * HD + j];
  float c1 = Cb[r * HD + j];
  float c2 = Cb[RR * HD + r * HD + j];
  __syncthreads();
  // layer 1
  float z[4];
  for (int g = 0; g < 4; ++g) z[g] = b1[g * HD + j];
  for (int k = 0; k < EMBD; ++k) {
    float xv = x[k];
    for (int g = 0; g < 4; ++g) z[g] += xv * Wi1[(size_t)k * (4*HD) + g * HD + j];
  }
  for (int k = 0; k < HD; ++k) {
    float hv = h1o[k];
    for (int g = 0; g < 4; ++g) z[g] += hv * Wh1[(size_t)k * (4*HD) + g * HD + j];
  }
  c1 = sigf(z[1]) * c1 + sigf(z[0]) * tanhf(z[2]);
  float h1v = sigf(z[3]) * tanhf(c1);
  h1n[j] = h1v;
  __syncthreads();
  // layer 2
  for (int g = 0; g < 4; ++g) z[g] = b2[g * HD + j];
  for (int k = 0; k < HD; ++k) {
    float xv = h1n[k];
    for (int g = 0; g < 4; ++g) z[g] += xv * Wi2[(size_t)k * (4*HD) + g * HD + j];
  }
  for (int k = 0; k < HD; ++k) {
    float hv = h2o[k];
    for (int g = 0; g < 4; ++g) z[g] += hv * Wh2[(size_t)k * (4*HD) + g * HD + j];
  }
  c2 = sigf(z[1]) * c2 + sigf(z[0]) * tanhf(z[2]);
  float h2v = sigf(z[3]) * tanhf(c2);
  h2n[j] = h2v;
  __syncthreads();
  // attention over enc rows of batch b
  if (j < SS) {
    float s = 0.f;
    const float* er = enc + ((size_t)b * SS + j) * HD;
    for (int h = 0; h < HD; ++h) s += h2n[h] * er[h];
    att[j] = s;
  }
  __syncthreads();
  if (j == 0) {
    float m = att[0];
    for (int s = 1; s < SS; ++s) m = fmaxf(m, att[s]);
    float sum = 0.f;
    for (int s = 0; s < SS; ++s) { att[s] = expf(att[s] - m); sum += att[s]; }
    float inv = 1.0f / sum;
    for (int s = 0; s < SS; ++s) att[s] *= inv;
  }
  __syncthreads();
  float ctx = 0.f;
  for (int s = 0; s < SS; ++s) ctx += att[s] * enc[((size_t)b * SS + s) * HD + j];
  // write back (hidden is intentionally NOT reordered by beam, per reference)
  Hb[r * HD + j]           = h1v;
  Hb[RR * HD + r * HD + j] = h2v;
  Cb[r * HD + j]           = c1;
  Cb[RR * HD + r * HD + j] = c2;
  A16[r * KK + j]          = (f16)h2v;
  A16[r * KK + HD + j]     = (f16)ctx;
  if (j == 0) {
    int bt = -1;
    if (t >= 3) {
      int w0 = seq_in[r * MAXLEN + t - 3];
      int w1 = seq_in[r * MAXLEN + t - 2];
      int w2 = seq_in[r * MAXLEN + t - 1];
      if (w0 == w1 && w1 == w2) bt = w0;
    }
    blk[r] = bt;
  }
}

// ============================================================
// 5) logits GEMM via WMMA f16->f32:  [32,512] x [512,NPAD]
//    8 waves/block, each wave owns one 16-col tile, both 16-row tiles.
// ============================================================
__global__ void __launch_bounds__(256) k_gemm(
    const f16* __restrict__ A16g, const f16* __restrict__ W16,
    const float* __restrict__ bo, const int* __restrict__ blk,
    float* __restrict__ logits) {
  __shared__ f16 As[RR * KK];            // 32 KB
  int tid = threadIdx.x;
  {
    const uint4* s = (const uint4*)A16g;
    uint4* d = (uint4*)As;
    for (int i = tid; i < (RR * KK) / 8; i += 256) d[i] = s[i];
  }
  __syncthreads();
  int w    = tid >> 5;
  int lane = tid & 31;
  int nt   = blockIdx.x * 8 + w;
  if (nt >= NT) return;                  // wave-uniform; EXEC stays full below
  int lo = lane & 15;
  int hi = lane >> 4;
  v8f acc0 = {0,0,0,0,0,0,0,0};
  v8f acc1 = {0,0,0,0,0,0,0,0};
  const uint4* bp = (const uint4*)(W16 + (size_t)nt * 16 * 512 + lane * 16);
  for (int kt = 0; kt < 16; ++kt) {
    VU bu; bu.q[0] = bp[0]; bu.q[1] = bp[1];
    bp += 64;                            // next K-tile (512 halfs)
    int k0 = kt * 32;
    int kA = k0 + hi * 8;
    int kB = k0 + 16 + hi * 8;
    VU a0, a1;
    a0.q[0] = *(const uint4*)&As[lo * KK + kA];
    a0.q[1] = *(const uint4*)&As[lo * KK + kB];
    a1.q[0] = *(const uint4*)&As[(16 + lo) * KK + kA];
    a1.q[1] = *(const uint4*)&As[(16 + lo) * KK + kB];
    acc0 = __builtin_amdgcn_wmma_f32_16x16x32_f16(false, a0.v, false, bu.v,
                                                  (short)0, acc0, false, false);
    acc1 = __builtin_amdgcn_wmma_f32_16x16x32_f16(false, a1.v, false, bu.v,
                                                  (short)0, acc1, false, false);
  }
  int n   = nt * 16 + lo;
  bool nv = (n < VSZ);
  float bias = nv ? bo[n] : 0.f;
  int rb = hi * 8;
  for (int r = 0; r < 8; ++r) {
    int row0 = rb + r;
    int row1 = 16 + rb + r;
    float v0 = nv ? (acc0[r] + bias) : NEGF;
    float v1 = nv ? (acc1[r] + bias) : NEGF;
    if (nv && n == blk[row0]) v0 = NEGF;
    if (nv && n == blk[row1]) v1 = NEGF;
    logits[(size_t)row0 * NPAD + n] = v0;
    logits[(size_t)row1 * NPAD + n] = v1;
  }
}

// ============================================================
// 6) per-row logsumexp
// ============================================================
__global__ void __launch_bounds__(256) k_rowstats(const float* __restrict__ logits,
                                                  float* __restrict__ logz) {
  int r = blockIdx.x, tid = threadIdx.x;
  __shared__ float red[256];
  const float* p = logits + (size_t)r * NPAD;
  float m = -3.4e38f;
  for (int n = tid; n < NPAD; n += 256) m = fmaxf(m, p[n]);
  red[tid] = m; __syncthreads();
  for (int s = 128; s > 0; s >>= 1) {
    if (tid < s) red[tid] = fmaxf(red[tid], red[tid + s]);
    __syncthreads();
  }
  m = red[0]; __syncthreads();
  float acc = 0.f;
  for (int n = tid; n < NPAD; n += 256) acc += expf(p[n] - m);
  red[tid] = acc; __syncthreads();
  for (int s = 128; s > 0; s >>= 1) {
    if (tid < s) red[tid] += red[tid + s];
    __syncthreads();
  }
  if (tid == 0) logz[r] = m + logf(red[0]);
}

// ============================================================
// 7) top-4 per batch over BEAMS*V scores
// ============================================================
__device__ __forceinline__ void top4_insert(float v, int fi, float tv[4], int ti[4]) {
  if (v > tv[3] || (v == tv[3] && fi < ti[3])) {
    int pos = 3;
    while (pos > 0 && (v > tv[pos-1] || (v == tv[pos-1] && fi < ti[pos-1]))) {
      tv[pos] = tv[pos-1]; ti[pos] = ti[pos-1]; --pos;
    }
    tv[pos] = v; ti[pos] = fi;
  }
}

__global__ void __launch_bounds__(256) k_topk(
    const float* __restrict__ logits, const float* __restrict__ logz,
    const float* __restrict__ bsin, float* __restrict__ bsout,
    int* __restrict__ topi) {
  int b = blockIdx.x, tid = threadIdx.x;
  float tv[4]; int ti[4];
  for (int i = 0; i < 4; ++i) { tv[i] = -3.4e38f; ti[i] = 0x7fffffff; }
  for (int j = tid; j < BEAMS * NPAD; j += 256) {
    int beam = j / NPAD;
    int n    = j - beam * NPAD;
    if (n >= VSZ) continue;
    int row  = b * BEAMS + beam;
    float v  = bsin[row] + logits[(size_t)row * NPAD + n] - logz[row];
    top4_insert(v, beam * VSZ + n, tv, ti);
  }
  __shared__ float sv[1024];
  __shared__ int   si[1024];
  for (int i = 0; i < 4; ++i) { sv[tid * 4 + i] = tv[i]; si[tid * 4 + i] = ti[i]; }
  __syncthreads();
  if (tid == 0) {
    float fv[4]; int fi[4];
    for (int i = 0; i < 4; ++i) { fv[i] = -3.4e38f; fi[i] = 0x7fffffff; }
    for (int e = 0; e < 1024; ++e) top4_insert(sv[e], si[e], fv, fi);
    for (int k = 0; k < 4; ++k) {
      bsout[b * BEAMS + k] = fv[k];
      topi [b * BEAMS + k] = fi[k];
    }
  }
}

// ============================================================
// 8) beam reorder / token append (seq ping-pong)
// ============================================================
__global__ void k_update(const int* __restrict__ seq_in, int* __restrict__ seq_out,
                         const int* __restrict__ topi, int t) {
  int gid = blockIdx.x * 256 + threadIdx.x;       // 0..1023
  if (gid >= BB * BEAMS * MAXLEN) return;
  int b = gid >> 7;
  int k = (gid >> 5) & 3;
  int i = gid & 31;
  int flat = topi[b * BEAMS + k];
  int bi   = flat / VSZ;
  int tok  = flat - bi * VSZ;
  int v = (i == t) ? tok : seq_in[(b * BEAMS + bi) * MAXLEN + i];
  seq_out[(b * BEAMS + k) * MAXLEN + i] = v;
}

// ============================================================
// 9) finalize: length penalty, emit seq[:,0,:] (as float) + scores
// ============================================================
__global__ void k_final(const int* __restrict__ seq, const float* __restrict__ bs,
                        float* __restrict__ out) {
  int tid = threadIdx.x;
  if (tid < BB * MAXLEN) {
    int b = tid >> 5, i = tid & 31;
    out[b * MAXLEN + i] = (float)seq[(b * BEAMS) * MAXLEN + i];
  }
  if (tid < BB) {
    int len = 0;
    for (int i = 0; i < MAXLEN; ++i)
      if (seq[(tid * BEAMS) * MAXLEN + i] != 0) ++len;
    float pen = (len + 1.0f) * 0.5f;      // LP = 1.0
    out[BB * MAXLEN + tid] = bs[tid * BEAMS] / pen;
  }
}

// ============================================================
extern "C" void kernel_launch(void* const* d_in, const int* in_sizes, int n_in,
                              void* d_out, int out_size, void* d_ws, size_t ws_size,
                              hipStream_t stream) {
  (void)in_sizes; (void)n_in; (void)out_size; (void)ws_size;
  const int*   src     = (const int*)  d_in[0];
  const float* E       = (const float*)d_in[1];
  const float* eWi1    = (const float*)d_in[2];
  const float* eWh1    = (const float*)d_in[3];
  const float* eb1     = (const float*)d_in[4];
  const float* eWi2    = (const float*)d_in[5];
  const float* eWh2    = (const float*)d_in[6];
  const float* eb2     = (const float*)d_in[7];
  const float* dWi1    = (const float*)d_in[8];
  const float* dWh1    = (const float*)d_in[9];
  const float* db1     = (const float*)d_in[10];
  const float* dWi2    = (const float*)d_in[11];
  const float* dWh2    = (const float*)d_in[12];
  const float* db2     = (const float*)d_in[13];
  const float* Wo      = (const float*)d_in[14];
  const float* bo      = (const float*)d_in[15];

  char* ws = (char*)d_ws;
  f16*   W16   = (f16*)  (ws + OFF_WO16);
  float* LOGI  = (float*)(ws + OFF_LOG);
  float* ENC   = (float*)(ws + OFF_ENC);
  float* Hb    = (float*)(ws + OFF_H);
  float* Cb    = (float*)(ws + OFF_C);
  f16*   A16   = (f16*)  (ws + OFF_A16);
  int*   SEQ0  = (int*)  (ws + OFF_SEQ0);
  int*   SEQ1  = (int*)  (ws + OFF_SEQ1);
  float* BS0   = (float*)(ws + OFF_BS0);
  float* BS1   = (float*)(ws + OFF_BS1);
  float* LOGZ  = (float*)(ws + OFF_LOGZ);
  int*   TOPI  = (int*)  (ws + OFF_TOPI);
  int*   BLK   = (int*)  (ws + OFF_BLK);

  // one-time (per call, deterministic) Wo -> f16 retile; L2-resident afterwards
  {
    int total = NT * 16 * 512;
    k_convert_wo<<<(total + 255) / 256, 256, 0, stream>>>(Wo, W16);
  }
  k_encoder<<<BB, 256, 0, stream>>>(src, E, eWi1, eWh1, eb1, eWi2, eWh2, eb2,
                                    ENC, Hb, Cb);
  k_init<<<4, 256, 0, stream>>>(SEQ0, BS0);

  for (int t = 1; t < MAXLEN; ++t) {
    int*   sin  = (t & 1) ? SEQ0 : SEQ1;
    int*   sout = (t & 1) ? SEQ1 : SEQ0;
    float* bin  = (t & 1) ? BS0  : BS1;
    float* bout = (t & 1) ? BS1  : BS0;
    k_step<<<RR, 256, 0, stream>>>(sin, E, dWi1, dWh1, db1, dWi2, dWh2, db2,
                                   ENC, Hb, Cb, A16, BLK, t);
    k_gemm<<<(NT + 7) / 8, 256, 0, stream>>>(A16, W16, bo, BLK, LOGI);
    k_rowstats<<<RR, 256, 0, stream>>>(LOGI, LOGZ);
    k_topk<<<BB, 256, 0, stream>>>(LOGI, LOGZ, bin, bout, TOPI);
    k_update<<<4, 256, 0, stream>>>(sin, sout, TOPI, t);
  }
  // t ran 1..31 (odd last) -> final state in SEQ1 / BS1
  k_final<<<1, 256, 0, stream>>>(SEQ1, BS1, (float*)d_out);
}
